// GIN_29618094473882
// MI455X (gfx1250) — compile-verified, matrix operand order checked by reading, not run on hardware
//
#include <hip/hip_runtime.h>

// ---- WMMA vector types (gfx1250, wave32) ----
typedef _Float16 v16h  __attribute__((ext_vector_type(16)));
typedef _Float16 half8 __attribute__((ext_vector_type(8)));
typedef float    v8f   __attribute__((ext_vector_type(8)));

#define LDA 136   // half-element pitch: 272B = 17*16 -> half8 chunks stay 16B-aligned, bank-spread
#define LDF 132   // float pitch: 528B = 33*16

// ---------------------------------------------------------------------------
// K0: float4 copy (agg := x)
// ---------------------------------------------------------------------------
__global__ void k_copy4(float4* __restrict__ dst, const float4* __restrict__ src, long n4) {
    long t = (long)blockIdx.x * blockDim.x + threadIdx.x;
    if (t < n4) dst[t] = src[t];
}

// ---------------------------------------------------------------------------
// K1/K4: edge scatter-add. 32 consecutive threads = one edge (512B coalesced
// gather); f32 atomics RMW in L2 (51MB accumulator << 192MB L2).
// ---------------------------------------------------------------------------
__global__ void k_scatter(const float* __restrict__ feat,
                          const int* __restrict__ src, const int* __restrict__ dst,
                          float* __restrict__ agg, long nWork) {
    long t = (long)blockIdx.x * blockDim.x + threadIdx.x;
    if (t >= nWork) return;
    long e = t >> 5;
    int  c = (int)(t & 31) << 2;
    int  s = src[e], d = dst[e];
    float4 v = *(const float4*)(feat + (long)s * 128 + c);
    float* p = agg + (long)d * 128 + c;
    unsafeAtomicAdd(p + 0, v.x);
    unsafeAtomicAdd(p + 1, v.y);
    unsafeAtomicAdd(p + 2, v.z);
    unsafeAtomicAdd(p + 3, v.w);
}

// ---------------------------------------------------------------------------
// WMMA fragment helper (ISA 7.12.2 layout): two contiguous half8 chunks,
// elems 0-7 -> K = k0 + 8*laneHi + [0..7], elems 8-15 -> +16.
// ---------------------------------------------------------------------------
__device__ __forceinline__ v16h load_frag(const _Float16* p, int k0) {
    half8 lo = *(const half8*)(p + k0);
    half8 hi = *(const half8*)(p + k0 + 16);
    return __builtin_shufflevector(lo, hi, 0,1,2,3,4,5,6,7,8,9,10,11,12,13,14,15);
}

#define WMMA_F16(a, b, c) \
    __builtin_amdgcn_wmma_f32_16x16x32_f16(false, (a), false, (b), (short)0, (c), false, false)

// ---------------------------------------------------------------------------
// K2/K5: O_f16[N,128] = relu(A_f32[N,128] @ W_f32[128,128] + b)
// Block = 256 thr (8 waves), 64-row x 128-col tile. Wave w owns n0=16w and
// sweeps 4 M-subtiles with resident B-fragments; results staged in LDS and
// written back as coalesced b128.
// ---------------------------------------------------------------------------
__global__ __launch_bounds__(256) void k_gemm_f32_h16(
    const float* __restrict__ A, const float* __restrict__ W,
    const float* __restrict__ bias, _Float16* __restrict__ O, int Nrows) {
    __shared__ __align__(16) _Float16 Wt[128 * LDA];
    __shared__ __align__(16) _Float16 As[64 * LDA];
    __shared__ __align__(16) _Float16 Os[64 * LDA];
    int tid = threadIdx.x;
    int m0  = blockIdx.x * 64;
    // W[128][128] f32 -> Wt[n][k] f16 (float4 reads)
    for (int idx4 = tid; idx4 < 128 * 32; idx4 += 256) {
        int k = idx4 >> 5, n4 = (idx4 & 31) << 2;
        float4 w = ((const float4*)W)[idx4];
        Wt[(n4 + 0) * LDA + k] = (_Float16)w.x;
        Wt[(n4 + 1) * LDA + k] = (_Float16)w.y;
        Wt[(n4 + 2) * LDA + k] = (_Float16)w.z;
        Wt[(n4 + 3) * LDA + k] = (_Float16)w.w;
    }
    // A tile 64x128 f32 -> f16
    for (int idx4 = tid; idx4 < 64 * 32; idx4 += 256) {
        int r = idx4 >> 5, c4 = (idx4 & 31) << 2;
        int row = m0 + r;
        float4 v = (row < Nrows) ? ((const float4*)(A + (long)row * 128))[idx4 & 31]
                                 : make_float4(0.f, 0.f, 0.f, 0.f);
        _Float16* p = As + r * LDA + c4;
        p[0] = (_Float16)v.x; p[1] = (_Float16)v.y;
        p[2] = (_Float16)v.z; p[3] = (_Float16)v.w;
    }
    __syncthreads();
    int lane = tid & 31, lanelo = lane & 15, laneHi = (lane >> 4) & 1;
    int n0 = (tid >> 5) * 16;
    float bval = bias[n0 + lanelo];
    const _Float16* pB = Wt + (n0 + lanelo) * LDA + 8 * laneHi;
    v16h b0 = load_frag(pB, 0), b1 = load_frag(pB, 32);
    v16h b2 = load_frag(pB, 64), b3 = load_frag(pB, 96);
#pragma unroll
    for (int mi = 0; mi < 4; ++mi) {
        const _Float16* pA = As + (mi * 16 + lanelo) * LDA + 8 * laneHi;
        v8f acc = { bval, bval, bval, bval, bval, bval, bval, bval };
        acc = WMMA_F16(load_frag(pA, 0),  b0, acc);
        acc = WMMA_F16(load_frag(pA, 32), b1, acc);
        acc = WMMA_F16(load_frag(pA, 64), b2, acc);
        acc = WMMA_F16(load_frag(pA, 96), b3, acc);
#pragma unroll
        for (int i = 0; i < 8; ++i) {
            float v = acc[i];
            v = v > 0.f ? v : 0.f;
            Os[(mi * 16 + i + 8 * laneHi) * LDA + n0 + lanelo] = (_Float16)v;
        }
    }
    __syncthreads();
    // coalesced writeback: one b128 (8 halves) per thread iteration
    for (int idx8 = tid; idx8 < 64 * 16; idx8 += 256) {
        int r = idx8 >> 4, c8 = (idx8 & 15) << 3;
        int row = m0 + r;
        if (row < Nrows)
            *(half8*)(O + (long)row * 128 + c8) = *(const half8*)(Os + r * LDA + c8);
    }
}

// ---------------------------------------------------------------------------
// K3: dual f32 output: O1 = O2 = relu(A_f16 @ W + b)   (H and agg2-init)
// ---------------------------------------------------------------------------
__global__ __launch_bounds__(256) void k_gemm_h16_dualf32(
    const _Float16* __restrict__ A, const float* __restrict__ W,
    const float* __restrict__ bias, float* __restrict__ O1, float* __restrict__ O2, int Nrows) {
    __shared__ __align__(16) _Float16 Wt[128 * LDA];
    __shared__ __align__(16) _Float16 As[64 * LDA];
    __shared__ __align__(16) float    Osf[64 * LDF];
    int tid = threadIdx.x;
    int m0  = blockIdx.x * 64;
    for (int idx4 = tid; idx4 < 128 * 32; idx4 += 256) {
        int k = idx4 >> 5, n4 = (idx4 & 31) << 2;
        float4 w = ((const float4*)W)[idx4];
        Wt[(n4 + 0) * LDA + k] = (_Float16)w.x;
        Wt[(n4 + 1) * LDA + k] = (_Float16)w.y;
        Wt[(n4 + 2) * LDA + k] = (_Float16)w.z;
        Wt[(n4 + 3) * LDA + k] = (_Float16)w.w;
    }
    for (int idx8 = tid; idx8 < 64 * 16; idx8 += 256) {   // A already f16: b128 loads
        int r = idx8 >> 4, c8 = (idx8 & 15) << 3;
        int row = m0 + r;
        half8 v;
        if (row < Nrows) v = *(const half8*)(A + (long)row * 128 + c8);
        else             v = (half8)(_Float16)0.f;
        *(half8*)(As + r * LDA + c8) = v;
    }
    __syncthreads();
    int lane = tid & 31, lanelo = lane & 15, laneHi = (lane >> 4) & 1;
    int n0 = (tid >> 5) * 16;
    float bval = bias[n0 + lanelo];
    const _Float16* pB = Wt + (n0 + lanelo) * LDA + 8 * laneHi;
    v16h b0 = load_frag(pB, 0), b1 = load_frag(pB, 32);
    v16h b2 = load_frag(pB, 64), b3 = load_frag(pB, 96);
#pragma unroll
    for (int mi = 0; mi < 4; ++mi) {
        const _Float16* pA = As + (mi * 16 + lanelo) * LDA + 8 * laneHi;
        v8f acc = { bval, bval, bval, bval, bval, bval, bval, bval };
        acc = WMMA_F16(load_frag(pA, 0),  b0, acc);
        acc = WMMA_F16(load_frag(pA, 32), b1, acc);
        acc = WMMA_F16(load_frag(pA, 64), b2, acc);
        acc = WMMA_F16(load_frag(pA, 96), b3, acc);
#pragma unroll
        for (int i = 0; i < 8; ++i) {
            float v = acc[i];
            v = v > 0.f ? v : 0.f;
            Osf[(mi * 16 + i + 8 * laneHi) * LDF + n0 + lanelo] = v;
        }
    }
    __syncthreads();
    // coalesced writeback: float4 per thread iteration, to both buffers
    for (int idx4 = tid; idx4 < 64 * 32; idx4 += 256) {
        int r = idx4 >> 5, c4 = (idx4 & 31) << 2;
        int row = m0 + r;
        if (row < Nrows) {
            float4 v = *(const float4*)(Osf + r * LDF + c4);
            *(float4*)(O1 + (long)row * 128 + c4) = v;
            *(float4*)(O2 + (long)row * 128 + c4) = v;
        }
    }
}

// ---------------------------------------------------------------------------
// K6: out[N,40] = log_softmax(A_f16[N,128] @ W_f32[128,40] + b) — fully fused.
// Block = 96 thr (3 waves), 64-row x 48-col tile (cols >=40 masked). Logits
// staged in LDS; per-row LSE computed in-block; final values written coalesced.
// ---------------------------------------------------------------------------
__global__ __launch_bounds__(96) void k_gemm_h16_lsm(
    const _Float16* __restrict__ A, const float* __restrict__ W,
    const float* __restrict__ bias, float* __restrict__ O, int Nrows) {
    __shared__ __align__(16) _Float16 Wt[48 * LDA];
    __shared__ __align__(16) _Float16 As[64 * LDA];
    __shared__ __align__(16) float    Osf[64 * 48];
    __shared__ float Ls[64];
    int tid = threadIdx.x;
    int m0  = blockIdx.x * 64;
    for (int idx = tid; idx < 48 * 128; idx += 96) {       // zero padded 48-col tile
        int k = idx & 127, n = idx >> 7;
        Wt[n * LDA + k] = (_Float16)0.f;
    }
    __syncthreads();
    for (int idx = tid; idx < 128 * 40; idx += 96) {       // W[128,40] transposed
        int k = idx / 40, n = idx % 40;
        Wt[n * LDA + k] = (_Float16)W[idx];
    }
    for (int idx8 = tid; idx8 < 64 * 16; idx8 += 96) {
        int r = idx8 >> 4, c8 = (idx8 & 15) << 3;
        int row = m0 + r;
        half8 v;
        if (row < Nrows) v = *(const half8*)(A + (long)row * 128 + c8);
        else             v = (half8)(_Float16)0.f;
        *(half8*)(As + r * LDA + c8) = v;
    }
    __syncthreads();
    int lane = tid & 31, lanelo = lane & 15, laneHi = (lane >> 4) & 1;
    int n0 = (tid >> 5) * 16;
    int n  = n0 + lanelo;
    float bval = (n < 40) ? bias[n] : 0.f;
    const _Float16* pB = Wt + n * LDA + 8 * laneHi;
    v16h b0 = load_frag(pB, 0), b1 = load_frag(pB, 32);
    v16h b2 = load_frag(pB, 64), b3 = load_frag(pB, 96);
#pragma unroll
    for (int mi = 0; mi < 4; ++mi) {
        const _Float16* pA = As + (mi * 16 + lanelo) * LDA + 8 * laneHi;
        v8f acc = { bval, bval, bval, bval, bval, bval, bval, bval };
        acc = WMMA_F16(load_frag(pA, 0),  b0, acc);
        acc = WMMA_F16(load_frag(pA, 32), b1, acc);
        acc = WMMA_F16(load_frag(pA, 64), b2, acc);
        acc = WMMA_F16(load_frag(pA, 96), b3, acc);
#pragma unroll
        for (int i = 0; i < 8; ++i)
            Osf[(mi * 16 + i + 8 * laneHi) * 48 + n] = acc[i];
    }
    __syncthreads();
    // per-row LSE (thread t < 64 owns row t; 40 values resident in LDS)
    if (tid < 64 && (m0 + tid) < Nrows) {
        const float* row = Osf + tid * 48;
        float m = row[0];
        for (int c = 1; c < 40; ++c) m = fmaxf(m, row[c]);
        float s = 0.f;
        for (int c = 0; c < 40; ++c) s += expf(row[c] - m);
        Ls[tid] = m + logf(s);
    }
    __syncthreads();
    // coalesced final write: out = logit - lse
    for (int idx = tid; idx < 64 * 40; idx += 96) {
        int r = idx / 40, c = idx % 40;
        int row = m0 + r;
        if (row < Nrows)
            O[(long)row * 40 + c] = Osf[r * 48 + c] - Ls[r];
    }
}

// ---------------------------------------------------------------------------
extern "C" void kernel_launch(void* const* d_in, const int* in_sizes, int n_in,
                              void* d_out, int out_size, void* d_ws, size_t ws_size,
                              hipStream_t stream) {
    const float* x   = (const float*)d_in[0];
    const int*   ei  = (const int*)d_in[1];
    const float* W1a = (const float*)d_in[2];
    const float* b1a = (const float*)d_in[3];
    const float* W1b = (const float*)d_in[4];
    const float* b1b = (const float*)d_in[5];
    const float* W2a = (const float*)d_in[6];
    const float* b2a = (const float*)d_in[7];
    const float* W2b = (const float*)d_in[8];
    const float* b2b = (const float*)d_in[9];

    const int  N  = in_sizes[0] / 128;      // 100000 nodes
    const long nE = in_sizes[1] / 2;        // 1600000 edges
    const int* src = ei;
    const int* dst = ei + nE;

    // workspace: agg (f32 N*128) | H (f32 N*128) | t16 (f16 N*128)  = 128MB
    float*     agg = (float*)d_ws;
    float*     H   = agg + (size_t)N * 128;
    _Float16*  t16 = (_Float16*)(H + (size_t)N * 128);
    float*     out = (float*)d_out;

    const long n4    = (long)N * 32;                 // float4 elements per feature tensor
    const long nWork = nE * 32;                      // (edge, 4-chan) tasks
    const int  gGemm = (N + 63) / 64;                // 64-row tiles

    // ---- layer 1 ----
    k_copy4<<<(unsigned)((n4 + 255) / 256), 256, 0, stream>>>((float4*)agg, (const float4*)x, n4);
    k_scatter<<<(unsigned)((nWork + 255) / 256), 256, 0, stream>>>(x, src, dst, agg, nWork);
    k_gemm_f32_h16<<<gGemm, 256, 0, stream>>>(agg, W1a, b1a, t16, N);
    k_gemm_h16_dualf32<<<gGemm, 256, 0, stream>>>(t16, W1b, b1b, H, agg, N);  // H = relu(.); agg := H

    // ---- layer 2 ----
    k_scatter<<<(unsigned)((nWork + 255) / 256), 256, 0, stream>>>(H, src, dst, agg, nWork);
    k_gemm_f32_h16<<<gGemm, 256, 0, stream>>>(agg, W2a, b2a, t16, N);
    k_gemm_h16_lsm<<<gGemm, 96, 0, stream>>>(t16, W2b, b2b, out, N);
}
